// LinearTrajectoryAttention_28982439314096
// MI455X (gfx1250) — 8-bit matrix operand order repaired and checked by reading, compiled, not run on hardware
//
#include <hip/hip_runtime.h>
#include <hip/hip_bf16.h>

#define D_MODEL   1024
#define NUM_HEADS 16
#define D_HEAD    64
#define BATCH     16
#define SEQ       8192
#define EPS_F     1e-6f

#define NCHUNKS   32
#define SUBROWS   64
#define SUBS      4
#define SLAB_PB   1040
#define TPB       80
#define WSCALE    16.0f
#define WINV      0.0625f

typedef __attribute__((ext_vector_type(8))) int   v8i;
typedef __attribute__((ext_vector_type(8))) float v8f;

#if __has_builtin(__builtin_amdgcn_cvt_pk_fp8_f32)
static __device__ __forceinline__ unsigned short pk_e4m3(float a, float b) {
  return (unsigned short)(__builtin_amdgcn_cvt_pk_fp8_f32(a, b, 0, false) & 0xffff);
}
static __device__ __forceinline__ unsigned char f2e4m3(float f) {
  return (unsigned char)(__builtin_amdgcn_cvt_pk_fp8_f32(f, 0.0f, 0, false) & 0xff);
}
#else
static __device__ __forceinline__ unsigned char f2e4m3(float f) {
  unsigned u = __float_as_uint(f);
  unsigned s = (u >> 24) & 0x80;
  float a = __builtin_fabsf(f);
  if (!(a < 448.0f)) return (unsigned char)(s | 0x7E);
  if (a < 0.0009765625f) return (unsigned char)s;
  int e = (int)((u >> 23) & 0xff) - 127;
  int eq = e < -6 ? -6 : e;
  float step = __uint_as_float((unsigned)(eq - 3 + 127) << 23);
  float q = rintf(a / step) * step;
  if (q > 448.0f) q = 448.0f;
  unsigned qu = __float_as_uint(q);
  int qe = (int)((qu >> 23) & 0xff) - 127;
  unsigned code;
  if (q == 0.0f) code = 0;
  else if (qe < -6) code = (unsigned)(q * 512.0f + 0.25f);
  else code = ((unsigned)(qe + 7) << 3) | ((qu >> 20) & 0x7);
  return (unsigned char)(s | code);
}
static __device__ __forceinline__ unsigned short pk_e4m3(float a, float b) {
  return (unsigned short)(f2e4m3(a) | ((unsigned short)f2e4m3(b) << 8));
}
#endif

static __device__ __forceinline__ float elu1(float x) {
  return x > 0.0f ? x + 1.0f : __expf(x);
}

static __device__ __forceinline__ v8i load_a8(const unsigned char* p) {
  union { v8i v; unsigned char b[32]; } u;
  __builtin_memcpy(&u.b[0],  p,      16);
  __builtin_memcpy(&u.b[16], p + 32, 16);
  return u.v;
}
static __device__ __forceinline__ v8i load_b8(const unsigned char* p) {
  union { v8i v; unsigned char b[32]; } u;
  __builtin_memcpy(&u.b[0],  p,      16);
  __builtin_memcpy(&u.b[16], p + 32, 16);
  return u.v;
}
static __device__ __forceinline__ v8f wmma_fp8(v8i a, v8i b, v8f c) {
  return __builtin_amdgcn_wmma_f32_16x16x64_fp8_fp8(a, b, (short)0, c, false, false);
}

__global__ void kw_cvt(const float* __restrict__ w, unsigned char* __restrict__ o, int n) {
  int i = blockIdx.x * blockDim.x + threadIdx.x;
  if (i < n) o[i] = f2e4m3(w[i] * WSCALE);
}

__global__ void k0_zero(float* __restrict__ p, long n) {
  long i = (long)blockIdx.x * blockDim.x + threadIdx.x;
  long stride = (long)gridDim.x * blockDim.x;
  for (; i < n; i += stride) p[i] = 0.0f;
}

__global__ __launch_bounds__(512)
void k1_proj_kv(const float* __restrict__ key, const float* __restrict__ value,
                const unsigned char* __restrict__ wk8,
                const unsigned char* __restrict__ wv8,
                const float* __restrict__ bk, const float* __restrict__ bv,
                float* __restrict__ KVg, float* __restrict__ Zg) {
  extern __shared__ unsigned char smem[];
  unsigned char* slabK = smem;
  unsigned char* slabV = slabK + SUBROWS * SLAB_PB;
  unsigned char* kpT   = slabV + SUBROWS * SLAB_PB;
  unsigned char* vpT   = kpT + NUM_HEADS * D_HEAD * TPB;

  const int b     = blockIdx.x;
  const int chunk = blockIdx.y;
  const int tid   = threadIdx.x;
  const int head  = tid >> 5;
  const int lane  = tid & 31;
  const int lhalf = (lane >> 4) & 1;
  const int l15   = lane & 15;
  const int klo   = lhalf * 8;
  const int kb    = lhalf * 16;

  unsigned char* kpw = kpT + head * (D_HEAD * TPB);
  unsigned char* vpw = vpT + head * (D_HEAD * TPB);

  v8f kv[4][4];
#pragma unroll
  for (int i = 0; i < 4; ++i)
#pragma unroll
    for (int j = 0; j < 4; ++j) kv[i][j] = (v8f){0, 0, 0, 0, 0, 0, 0, 0};

  const long tbase = (long)chunk * (SEQ / NCHUNKS);

#pragma unroll 1
  for (int sc = 0; sc < SUBS; ++sc) {
    __syncthreads();
    {
      const long t0 = tbase + (long)sc * SUBROWS;
      const float* kb_ = key   + ((long)b * SEQ + t0) * D_MODEL;
      const float* vb_ = value + ((long)b * SEQ + t0) * D_MODEL;
      for (int idx = tid; idx < SUBROWS * (D_MODEL / 2); idx += 512) {
        int r = idx >> 9, dp = idx & (D_MODEL / 2 - 1);
        const float* kp = kb_ + (long)r * D_MODEL + 2 * dp;
        const float* vp = vb_ + (long)r * D_MODEL + 2 * dp;
        ((unsigned short*)(slabK + r * SLAB_PB))[dp] =
            pk_e4m3(__builtin_nontemporal_load(kp), __builtin_nontemporal_load(kp + 1));
        ((unsigned short*)(slabV + r * SLAB_PB))[dp] =
            pk_e4m3(__builtin_nontemporal_load(vp), __builtin_nontemporal_load(vp + 1));
      }
    }
    __syncthreads();

#pragma unroll 1
    for (int proj = 0; proj < 2; ++proj) {
      const unsigned char* slab = proj ? slabV : slabK;
      const unsigned char* wgt  = proj ? wv8 : wk8;
      const float* bias_g       = proj ? bv : bk;
      unsigned char* stg        = proj ? vpw : kpw;

#pragma unroll 1
      for (int np = 0; np < 2; ++np) {
        v8f c[4][2];
#pragma unroll
        for (int mt = 0; mt < 4; ++mt)
#pragma unroll
          for (int nt = 0; nt < 2; ++nt) c[mt][nt] = (v8f){0, 0, 0, 0, 0, 0, 0, 0};

        const unsigned char* wrow0 =
            wgt + (head * 64 + (np * 2 + 0) * 16 + l15) * D_MODEL + kb;
        const unsigned char* wrow1 =
            wgt + (head * 64 + (np * 2 + 1) * 16 + l15) * D_MODEL + kb;
        v8i b0 = load_b8(wrow0);
        v8i b1 = load_b8(wrow1);

#pragma unroll 2
        for (int d0 = 0; d0 < D_MODEL; d0 += 64) {
          v8i nb0 = load_b8(wrow0 + d0 + 64);
          v8i nb1 = load_b8(wrow1 + d0 + 64);
#pragma unroll
          for (int mt = 0; mt < 4; ++mt) {
            v8i a = load_a8(slab + (mt * 16 + l15) * SLAB_PB + d0 + kb);
            c[mt][0] = wmma_fp8(a, b0, c[mt][0]);
            c[mt][1] = wmma_fp8(a, b1, c[mt][1]);
          }
          b0 = nb0;
          b1 = nb1;
        }

#pragma unroll
        for (int nt = 0; nt < 2; ++nt) {
          int ntg = np * 2 + nt;
          int col = head * 64 + ntg * 16 + l15;
          float bias = bias_g[col];
          float z = 0.0f;
          unsigned short* row16 = (unsigned short*)(stg + (ntg * 16 + l15) * TPB);
#pragma unroll
          for (int mt = 0; mt < 4; ++mt)
#pragma unroll
            for (int r = 0; r < 8; r += 2) {
              float v0 = c[mt][nt][r] * WINV + bias;
              float v1 = c[mt][nt][r + 1] * WINV + bias;
              if (proj == 0) { v0 = elu1(v0); v1 = elu1(v1); z += v0 + v1; }
              row16[(mt * 16 + lhalf * 8 + r) >> 1] = pk_e4m3(v0, v1);
            }
          if (proj == 0)
            unsafeAtomicAdd(&Zg[(b * NUM_HEADS + head) * D_HEAD + ntg * 16 + l15], z);
        }
      }
    }

#pragma unroll
    for (int i = 0; i < 4; ++i) {
      v8i ak = load_a8(kpw + (i * 16 + l15) * TPB + kb);
#pragma unroll
      for (int j = 0; j < 4; ++j) {
        v8i bvv = load_b8(vpw + (j * 16 + l15) * TPB + kb);
        kv[i][j] = wmma_fp8(ak, bvv, kv[i][j]);
      }
    }
  }

#pragma unroll
  for (int i = 0; i < 4; ++i)
#pragma unroll
    for (int j = 0; j < 4; ++j)
#pragma unroll
      for (int r = 0; r < 8; ++r) {
        int dk = i * 16 + lhalf * 8 + r;
        int de = j * 16 + l15;
        unsafeAtomicAdd(&KVg[((long)(b * NUM_HEADS + head) * D_HEAD + dk) * D_HEAD + de],
                        kv[i][j][r]);
      }
}

__global__ __launch_bounds__(256)
void k2_finish(const float* __restrict__ query, const float* __restrict__ Wq,
               const float* __restrict__ bq, const float* __restrict__ KVg,
               const float* __restrict__ Zg, const float* __restrict__ Wo,
               const float* __restrict__ bo, float* __restrict__ out) {
  __shared__ float qf[D_MODEL];
  __shared__ float attn[D_MODEL];
  const int b = blockIdx.x, tid = threadIdx.x;

  for (int o = tid; o < D_MODEL; o += 256) {
    float acc = bq[o];
    const float* q = query + (long)b * D_MODEL;
    const float* w = Wq + (long)o * D_MODEL;
    for (int d = 0; d < D_MODEL; ++d) acc = __fmaf_rn(q[d], w[d], acc);
    qf[o] = elu1(acc);
  }
  __syncthreads();
  for (int idx = tid; idx < D_MODEL; idx += 256) {
    int h = idx >> 6, e = idx & 63;
    const float* kvp = KVg + (long)(b * NUM_HEADS + h) * D_HEAD * D_HEAD + e;
    const float* zp  = Zg + (long)(b * NUM_HEADS + h) * D_HEAD;
    const float* qh  = qf + h * D_HEAD;
    float num = 0.0f, den = 0.0f;
    for (int d = 0; d < D_HEAD; ++d) {
      num = __fmaf_rn(qh[d], kvp[d * D_HEAD], num);
      den = __fmaf_rn(qh[d], zp[d], den);
    }
    attn[idx] = num / (den + EPS_F);
  }
  __syncthreads();
  for (int o = tid; o < D_MODEL; o += 256) {
    float acc = bo[o];
    const float* w = Wo + (long)o * D_MODEL;
    for (int m = 0; m < D_MODEL; ++m) acc = __fmaf_rn(attn[m], w[m], acc);
    out[(long)b * D_MODEL + o] = acc;
  }
}

extern "C" void kernel_launch(void* const* d_in, const int* in_sizes, int n_in,
                              void* d_out, int out_size, void* d_ws, size_t ws_size,
                              hipStream_t stream) {
  const float* query = (const float*)d_in[0];
  const float* key   = (const float*)d_in[1];
  const float* value = (const float*)d_in[2];
  const float* Wq    = (const float*)d_in[3];
  const float* bq    = (const float*)d_in[4];
  const float* Wk    = (const float*)d_in[5];
  const float* bk    = (const float*)d_in[6];
  const float* Wv    = (const float*)d_in[7];
  const float* bv    = (const float*)d_in[8];
  const float* Wo    = (const float*)d_in[9];
  const float* bo    = (const float*)d_in[10];
  float* out = (float*)d_out;

  char* ws = (char*)d_ws;
  unsigned char* wk8 = (unsigned char*)ws;
  unsigned char* wv8 = (unsigned char*)(ws + (size_t)D_MODEL * D_MODEL);
  float* KVg = (float*)(ws + 2 * (size_t)D_MODEL * D_MODEL);
  float* Zg  = KVg + (size_t)BATCH * NUM_HEADS * D_HEAD * D_HEAD;

  const int nW = D_MODEL * D_MODEL;
  kw_cvt<<<(nW + 255) / 256, 256, 0, stream>>>(Wk, wk8, nW);
  kw_cvt<<<(nW + 255) / 256, 256, 0, stream>>>(Wv, wv8, nW);

  long nz = (long)BATCH * NUM_HEADS * D_HEAD * D_HEAD +
            (long)BATCH * NUM_HEADS * D_HEAD;
  k0_zero<<<512, 256, 0, stream>>>(KVg, nz);

  size_t lds = (size_t)(2 * SUBROWS * SLAB_PB + 2 * NUM_HEADS * D_HEAD * TPB);
  dim3 grid(BATCH, NCHUNKS);
  k1_proj_kv<<<grid, 512, lds, stream>>>(key, value, wk8, wv8, bk, bv, KVg, Zg);

  k2_finish<<<BATCH, 256, 0, stream>>>(query, Wq, bq, KVg, Zg, Wo, bo, out);
}
